// FallbackSSMKernel_14611478741277
// MI455X (gfx1250) — matrix operand order repaired and checked by reading, not compiled, not run on hardware
//
#include <hip/hip_runtime.h>
#include <hip/hip_bf16.h>
#include <stdint.h>

// ---------------------------------------------------------------------------
// Mamba-2 chunked SSM for MI455X (gfx1250), bf16 WMMA pipeline.
//   N_HEADS=64, HEAD_DIM=32, D_STATE=128, CHUNK=256, b=2, l=2048, d=2048
// dt/dA are per-head constants => all segsum decays are analytic exp(dA*k).
// GEMMs: bf16-prepacked operands + async global->LDS staging (double buffer),
// 64x64 wave tiles (16 WMMA per K-step per wave).
// ---------------------------------------------------------------------------

typedef __attribute__((ext_vector_type(16))) __bf16 v16bf;
typedef __attribute__((ext_vector_type(8)))  float  v8f;

// exact parameter types of __builtin_amdgcn_global_load_async_to_lds_b128:
//   (v4i __device__*, v4i __shared__*, imm offset, imm cpol)
typedef int v4i_vs __attribute__((vector_size(16)));
typedef v4i_vs as1_v4i __attribute__((address_space(1)));
typedef v4i_vs as3_v4i __attribute__((address_space(3)));

#define SSM_DT_CONST 0.1f

#if __has_builtin(__builtin_amdgcn_global_load_async_to_lds_b128) && \
    __has_builtin(__builtin_amdgcn_s_wait_asynccnt)
#define SSM_ASYNC_LDS 1
#else
#define SSM_ASYNC_LDS 0
#endif

__device__ __forceinline__ float softplus_f(float v) { return log1pf(__expf(v)); }

__device__ __forceinline__ v8f wmma_bf16(v16bf a, v16bf b, v8f c) {
  return __builtin_amdgcn_wmma_f32_16x16x32_bf16(false, a, false, b, (short)0, c,
                                                 false, false);
}

// 16-byte global->LDS copy: async DMA on CDNA5 (ASYNCcnt), else direct b128.
__device__ __forceinline__ void cp16_g2l(const __bf16* g, __bf16* l) {
#if SSM_ASYNC_LDS
  __builtin_amdgcn_global_load_async_to_lds_b128(
      (as1_v4i*)(uintptr_t)g,
      (as3_v4i*)(unsigned)(uintptr_t)l, 0, 0);
#else
  *reinterpret_cast<uint4*>(l) = *reinterpret_cast<const uint4*>(g);
#endif
}
__device__ __forceinline__ void cp16_wait() {
#if SSM_ASYNC_LDS
  __builtin_amdgcn_s_wait_asynccnt(0);
#endif
}

// A-matrix fragment (16x32 bf16), CDNA5 wave32 layout:
//   lanes 0-15: m=lane, VGPR j<4 -> k=2j ; j>=4 -> k=16+2(j-4); lanes16-31: +8.
template <int W>
__device__ __forceinline__ v16bf frag_A(const __bf16 (*L)[W], int mrow, int half) {
  v16bf f;
#pragma unroll
  for (int j = 0; j < 8; ++j) {
    int kk = (j < 4) ? (half * 8 + 2 * j) : (16 + half * 8 + 2 * (j - 4));
    f[2 * j]     = L[mrow][kk];
    f[2 * j + 1] = L[mrow][kk + 1];
  }
  return f;
}

// B-matrix fragment (32x16 bf16): same K order as frag_A (j<4 -> k=8*half+2j,
// j>=4 -> k=16+8*half+2(j-4)); lane&15 = column n. LDS stored [n][k].
template <int W>
__device__ __forceinline__ v16bf frag_B(const __bf16 (*L)[W], int nrow, int half) {
  v16bf f;
#pragma unroll
  for (int j = 0; j < 8; ++j) {
    int kk = (j < 4) ? (half * 8 + 2 * j) : (16 + half * 8 + 2 * (j - 4));
    f[2 * j]     = L[nrow][kk];
    f[2 * j + 1] = L[nrow][kk + 1];
  }
  return f;
}

// ---------------------------------------------------------------------------
// fp32 -> bf16 prepack (halves GEMM global traffic; enables async staging)
// ---------------------------------------------------------------------------
__global__ __launch_bounds__(256) void pack_bf16(const float* __restrict__ src,
                                                 __bf16* __restrict__ dst, int n) {
  int i = (blockIdx.x * 256 + threadIdx.x) * 4;
  if (i < n) {
    float4 v = *reinterpret_cast<const float4*>(src + i);
    dst[i + 0] = (__bf16)v.x; dst[i + 1] = (__bf16)v.y;
    dst[i + 2] = (__bf16)v.z; dst[i + 3] = (__bf16)v.w;
  }
}

// ---------------------------------------------------------------------------
// C[M,N] = A[M,K] @ B[N,K]^T (bf16 in, f32 WMMA accum, f32 out).
// 256 threads / 8 waves; block tile 128(M) x 256(N); K-step 32.
// Waves 2(m) x 4(n): each wave owns a 64x64 tile = 4x4 16x16 frags
// (16 WMMA per K-step per wave, 8 frag ds-loads). Double-buffered LDS with
// async global->LDS DMA staging (ASYNCcnt).
// ---------------------------------------------------------------------------
__global__ __launch_bounds__(256) void wmma_gemm_abt_bf(
    const __bf16* __restrict__ A, const __bf16* __restrict__ Bm,
    float* __restrict__ C, int M, int N, int K,
    long long sA, long long sB, long long sC) {
  __shared__ alignas(16) __bf16 la[2][128][40];
  __shared__ alignas(16) __bf16 lb[2][256][40];
  A  += (long long)blockIdx.z * sA;
  Bm += (long long)blockIdx.z * sB;
  C  += (long long)blockIdx.z * sC;
  const int t = threadIdx.x, lane = t & 31, half = lane >> 4, l16 = lane & 15;
  const int wave = t >> 5, wm = wave & 1, wn = wave >> 1;
  const int m0 = blockIdx.x * 128, n0 = blockIdx.y * 256;
  const int rr = t >> 2;              // 0..63: base row, 16B segment per thread
  const int sg = (t & 3) * 8;

  v8f acc[4][4];
#pragma unroll
  for (int mi = 0; mi < 4; ++mi)
#pragma unroll
    for (int ni = 0; ni < 4; ++ni)
#pragma unroll
      for (int r = 0; r < 8; ++r) acc[mi][ni][r] = 0.f;

  // prologue: stage k0 = 0 into buffer 0
#pragma unroll
  for (int q = 0; q < 2; ++q)
    cp16_g2l(A + (size_t)(m0 + rr + q * 64) * K + sg, &la[0][rr + q * 64][sg]);
#pragma unroll
  for (int q = 0; q < 4; ++q)
    cp16_g2l(Bm + (size_t)(n0 + rr + q * 64) * K + sg, &lb[0][rr + q * 64][sg]);
  cp16_wait();
  __syncthreads();

  int buf = 0;
  for (int k0 = 0; k0 < K; k0 += 32) {
    const bool more = (k0 + 32) < K;
    if (more) {  // async-stage next K-tile into the other buffer
      const int kn = k0 + 32, nb = buf ^ 1;
#pragma unroll
      for (int q = 0; q < 2; ++q)
        cp16_g2l(A + (size_t)(m0 + rr + q * 64) * K + kn + sg, &la[nb][rr + q * 64][sg]);
#pragma unroll
      for (int q = 0; q < 4; ++q)
        cp16_g2l(Bm + (size_t)(n0 + rr + q * 64) * K + kn + sg, &lb[nb][rr + q * 64][sg]);
    }
    v16bf af[4], bf[4];
#pragma unroll
    for (int mi = 0; mi < 4; ++mi) af[mi] = frag_A(la[buf], wm * 64 + mi * 16 + l16, half);
#pragma unroll
    for (int ni = 0; ni < 4; ++ni) bf[ni] = frag_B(lb[buf], wn * 64 + ni * 16 + l16, half);
#pragma unroll
    for (int mi = 0; mi < 4; ++mi)
#pragma unroll
      for (int ni = 0; ni < 4; ++ni)
        acc[mi][ni] = wmma_bf16(af[mi], bf[ni], acc[mi][ni]);
    if (more) {
      cp16_wait();
      __syncthreads();
    }
    buf ^= 1;
  }

#pragma unroll
  for (int mi = 0; mi < 4; ++mi)
#pragma unroll
    for (int ni = 0; ni < 4; ++ni) {
      const int mt = m0 + wm * 64 + mi * 16 + 8 * half;  // D: m = r + 8*half
      const int nt = n0 + wn * 64 + ni * 16 + l16;       //    n = lane&15
      float* Cp = C + (size_t)mt * N + nt;
#pragma unroll
      for (int r = 0; r < 8; ++r)
        Cp[(size_t)r * N] = acc[mi][ni][r];
    }
}

// ---------------------------------------------------------------------------
// Per (b,c,h): Y_diag = (tril(G) .* exp(dA*(l-s))) @ Xdt   [256x256 @ 256x32]
//              states = (Xdt .* exp(dA*(255-l)))^T @ B     [32x256 @ 256x128]
// ---------------------------------------------------------------------------
__global__ __launch_bounds__(256) void intra_chunk(
    const float* __restrict__ xz, const float* __restrict__ G,
    const float* __restrict__ Bmat, const float* __restrict__ A_log,
    const float* __restrict__ dt_bias, float* __restrict__ Y,
    float* __restrict__ st) {
  __shared__ __bf16 lg[256][36];
  __shared__ __bf16 lx[128][36];
  const int t = threadIdx.x, lane = t & 31, half = lane >> 4, l16 = lane & 15;
  const int wave = t >> 5;
  const int bch = blockIdx.x;
  const int h = bch & 63, c = (bch >> 6) & 7, b = bch >> 9;
  const float dt = softplus_f(SSM_DT_CONST + dt_bias[h]);
  const float dA = dt * (-__expf(A_log[h]));
  const size_t row0 = (size_t)b * 2048 + (size_t)c * 256;
  const size_t gbase = (size_t)(b * 8 + c) * 65536;

  // ---- phase 1: Y_diag (wave w owns rows [32w,32w+32), 2x2 frags)
  v8f acc1[2][2];
#pragma unroll
  for (int mi = 0; mi < 2; ++mi)
#pragma unroll
    for (int ni = 0; ni < 2; ++ni)
#pragma unroll
      for (int r = 0; r < 8; ++r) acc1[mi][ni][r] = 0.f;

  for (int s0 = 0; s0 < 256; s0 += 32) {
    __syncthreads();
#pragma unroll
    for (int i = 0; i < 8; ++i) {  // masked/decayed G tile 256x32
      int idx = t + i * 256;
      int l = idx >> 3, s4 = (idx & 7) * 4;
      float4 v = *reinterpret_cast<const float4*>(G + gbase + (size_t)l * 256 + s0 + s4);
      float vv[4] = {v.x, v.y, v.z, v.w};
#pragma unroll
      for (int e = 0; e < 4; ++e) {
        int s = s0 + s4 + e;
        float w = (s <= l) ? vv[e] * __expf(dA * (float)(l - s)) : 0.f;
        lg[l][s4 + e] = (__bf16)w;
      }
    }
#pragma unroll
    for (int i = 0; i < 4; ++i) {  // Xdt^T tile [p][s] 32x32
      int idx = t + i * 256;
      int p = idx >> 5, sl = idx & 31;
      size_t row = row0 + s0 + sl;
      lx[p][sl] = (__bf16)(xz[row * 4096 + h * 32 + p] * dt);
    }
    __syncthreads();
    v16bf af[2], bf[2];
#pragma unroll
    for (int mi = 0; mi < 2; ++mi) af[mi] = frag_A(lg, wave * 32 + mi * 16 + l16, half);
#pragma unroll
    for (int ni = 0; ni < 2; ++ni) bf[ni] = frag_B(lx, ni * 16 + l16, half);
#pragma unroll
    for (int mi = 0; mi < 2; ++mi)
#pragma unroll
      for (int ni = 0; ni < 2; ++ni)
        acc1[mi][ni] = wmma_bf16(af[mi], bf[ni], acc1[mi][ni]);
  }
#pragma unroll
  for (int mi = 0; mi < 2; ++mi)
#pragma unroll
    for (int ni = 0; ni < 2; ++ni)
#pragma unroll
      for (int r = 0; r < 8; ++r) {
        int l = wave * 32 + mi * 16 + 8 * half + r;
        int p = ni * 16 + l16;
        Y[((row0 + l) * 64 + h) * 32 + p] = acc1[mi][ni][r];
      }

  // ---- phase 2: states[p,n] (wave: m-tile = w>>2, n-tiles = (w&3)*2+{0,1})
  v8f acc2[2];
#pragma unroll
  for (int ni = 0; ni < 2; ++ni)
#pragma unroll
    for (int r = 0; r < 8; ++r) acc2[ni][r] = 0.f;

  for (int l0 = 0; l0 < 256; l0 += 32) {
    __syncthreads();
#pragma unroll
    for (int i = 0; i < 4; ++i) {  // A-op: [p][l] = xdt * exp(dA*(255-l))
      int idx = t + i * 256;
      int p = idx >> 5, ll = idx & 31;
      int lgl = l0 + ll;
      size_t row = row0 + lgl;
      float val = xz[row * 4096 + h * 32 + p] * dt * __expf(dA * (float)(255 - lgl));
      lg[p][ll] = (__bf16)val;
    }
#pragma unroll
    for (int i = 0; i < 4; ++i) {  // B-op: [n][l] from Bmat rows (n contiguous)
      int idx = t + i * 256;
      int ll = idx >> 5, n4 = (idx & 31) * 4;
      float4 v = *reinterpret_cast<const float4*>(Bmat + (row0 + l0 + ll) * 128 + n4);
      lx[n4 + 0][ll] = (__bf16)v.x; lx[n4 + 1][ll] = (__bf16)v.y;
      lx[n4 + 2][ll] = (__bf16)v.z; lx[n4 + 3][ll] = (__bf16)v.w;
    }
    __syncthreads();
    v16bf af = frag_A(lg, (wave >> 2) * 16 + l16, half);
    v16bf bf0 = frag_B(lx, ((wave & 3) * 2 + 0) * 16 + l16, half);
    v16bf bf1 = frag_B(lx, ((wave & 3) * 2 + 1) * 16 + l16, half);
    acc2[0] = wmma_bf16(af, bf0, acc2[0]);
    acc2[1] = wmma_bf16(af, bf1, acc2[1]);
  }
  const size_t sbase = (size_t)((b * 8 + c) * 64 + h) * 32 * 128;
#pragma unroll
  for (int ni = 0; ni < 2; ++ni)
#pragma unroll
    for (int r = 0; r < 8; ++r) {
      int p = (wave >> 2) * 16 + 8 * half + r;
      int n = ((wave & 3) * 2 + ni) * 16 + l16;
      st[sbase + (size_t)p * 128 + n] = acc2[ni][r];
    }
}

// ---------------------------------------------------------------------------
// Chunk recurrence (8 steps; chunk decay exp(256*dA[h]) is constant).
// ---------------------------------------------------------------------------
__global__ __launch_bounds__(256) void chunk_scan(
    const float* __restrict__ st, const float* __restrict__ A_log,
    const float* __restrict__ dt_bias, float* __restrict__ pv) {
  int idx = blockIdx.x * 256 + threadIdx.x;  // 2*64*32*128 = 524288
  int n = idx & 127, p = (idx >> 7) & 31, h = (idx >> 12) & 63, b = idx >> 18;
  float dt = softplus_f(SSM_DT_CONST + dt_bias[h]);
  float dA = dt * (-__expf(A_log[h]));
  float cd = __expf(dA * 256.f);
  float run = 0.f;
#pragma unroll
  for (int c = 0; c < 8; ++c) {
    size_t off = (size_t)(((b * 8 + c) * 64 + h) * 32 + p) * 128 + n;
    pv[off] = run;
    run = run * cd + st[off];
  }
}

// ---------------------------------------------------------------------------
// Per (b,c,h): Y_off = exp(dA*(l+1)) * (C @ prev_states^T)  [256x128 @ 128x32]
// Fused epilogue: y = (Y_off + Y_diag + D*x) * sigmoid(z)  -> bf16 for GEMM-2.
// ---------------------------------------------------------------------------
__global__ __launch_bounds__(256) void inter_chunk(
    const float* __restrict__ xz, const float* __restrict__ Cmat,
    const float* __restrict__ pv, const float* __restrict__ A_log,
    const float* __restrict__ dt_bias, const float* __restrict__ D_param,
    const float* __restrict__ Y, __bf16* __restrict__ yg) {
  __shared__ __bf16 la[256][36];
  __shared__ __bf16 lb[32][36];
  const int t = threadIdx.x, lane = t & 31, half = lane >> 4, l16 = lane & 15;
  const int wave = t >> 5;
  const int bch = blockIdx.x;
  const int h = bch & 63, c = (bch >> 6) & 7, b = bch >> 9;
  const float dt = softplus_f(SSM_DT_CONST + dt_bias[h]);
  const float dA = dt * (-__expf(A_log[h]));
  const float Dh = D_param[h];
  const size_t row0 = (size_t)b * 2048 + (size_t)c * 256;
  const size_t pbase = (size_t)((b * 8 + c) * 64 + h) * 32 * 128;

  v8f acc[2][2];
#pragma unroll
  for (int mi = 0; mi < 2; ++mi)
#pragma unroll
    for (int ni = 0; ni < 2; ++ni)
#pragma unroll
      for (int r = 0; r < 8; ++r) acc[mi][ni][r] = 0.f;

  for (int k0 = 0; k0 < 128; k0 += 32) {
    __syncthreads();
#pragma unroll
    for (int i = 0; i < 8; ++i) {  // C tile 256x32
      int idx = t + i * 256;
      int l = idx >> 3, k4 = (idx & 7) * 4;
      float4 v = *reinterpret_cast<const float4*>(Cmat + (row0 + l) * 128 + k0 + k4);
      la[l][k4 + 0] = (__bf16)v.x; la[l][k4 + 1] = (__bf16)v.y;
      la[l][k4 + 2] = (__bf16)v.z; la[l][k4 + 3] = (__bf16)v.w;
    }
    {  // prev_states tile [p][n] 32x32
      int p = t >> 3, k4 = (t & 7) * 4;
      float4 v = *reinterpret_cast<const float4*>(pv + pbase + (size_t)p * 128 + k0 + k4);
      lb[p][k4 + 0] = (__bf16)v.x; lb[p][k4 + 1] = (__bf16)v.y;
      lb[p][k4 + 2] = (__bf16)v.z; lb[p][k4 + 3] = (__bf16)v.w;
    }
    __syncthreads();
    v16bf af[2], bf[2];
#pragma unroll
    for (int mi = 0; mi < 2; ++mi) af[mi] = frag_A(la, wave * 32 + mi * 16 + l16, half);
#pragma unroll
    for (int ni = 0; ni < 2; ++ni) bf[ni] = frag_B(lb, ni * 16 + l16, half);
#pragma unroll
    for (int mi = 0; mi < 2; ++mi)
#pragma unroll
      for (int ni = 0; ni < 2; ++ni)
        acc[mi][ni] = wmma_bf16(af[mi], bf[ni], acc[mi][ni]);
  }

#pragma unroll
  for (int mi = 0; mi < 2; ++mi)
#pragma unroll
    for (int ni = 0; ni < 2; ++ni)
#pragma unroll
      for (int r = 0; r < 8; ++r) {
        int l = wave * 32 + mi * 16 + 8 * half + r;
        int p = ni * 16 + l16;
        size_t row = row0 + l;
        float x = xz[row * 4096 + h * 32 + p];
        float z = xz[row * 4096 + 2048 + h * 32 + p];
        float y = acc[mi][ni][r] * __expf(dA * (float)(l + 1)) +
                  Y[(row * 64 + h) * 32 + p] + Dh * x;
        yg[row * 2048 + h * 32 + p] = (__bf16)(y * (1.f / (1.f + __expf(-z))));
      }
}

// ---------------------------------------------------------------------------
extern "C" void kernel_launch(void* const* d_in, const int* in_sizes, int n_in,
                              void* d_out, int out_size, void* d_ws, size_t ws_size,
                              hipStream_t stream) {
  (void)in_sizes; (void)n_in; (void)out_size; (void)ws_size;
  const float* hidden = (const float*)d_in[0];
  const float* W_in   = (const float*)d_in[1];
  const float* W_out  = (const float*)d_in[2];
  const float* A_log  = (const float*)d_in[3];
  const float* D_par  = (const float*)d_in[4];
  const float* dt_b   = (const float*)d_in[5];
  const float* Bmat   = (const float*)d_in[6];
  const float* Cmat   = (const float*)d_in[7];
  float* out = (float*)d_out;

  // --- workspace layout (byte offsets; aliased regions noted) ---
  char* base = (char*)d_ws;
  float*  xz      = (float*)(base);                    //  64 MB
  float*  G       = (float*)(base + 67108864);         //   4 MB
  float*  Y       = (float*)(base + 71303168);         //  32 MB (Y_diag)
  float*  st      = (float*)(base + 104857600);        //  16 MB (alias Win_bf)
  __bf16* Win_bf  = (__bf16*)(base + 104857600);       //  dead after GEMM-1
  float*  pv      = (float*)(base + 121634816);        //  16 MB (alias hid_bf)
  __bf16* hid_bf  = (__bf16*)(base + 121634816);       //  dead after GEMM-1
  __bf16* Wout_bf = (__bf16*)(base + 138412032);       //   8 MB
  __bf16* Bm_bf   = (__bf16*)(base + 146800640);       //   1 MB
  __bf16* Cm_bf   = (__bf16*)(base + 147849216);       //   1 MB
  __bf16* yg_bf   = (__bf16*)(base + 148897792);       //  16 MB

  // 0) prepack fp32 -> bf16
  pack_bf16<<<8192, 256, 0, stream>>>(hidden, hid_bf, 8388608);
  pack_bf16<<<8192, 256, 0, stream>>>(W_in, Win_bf, 8388608);
  pack_bf16<<<4096, 256, 0, stream>>>(W_out, Wout_bf, 4194304);
  pack_bf16<<<512, 256, 0, stream>>>(Bmat, Bm_bf, 524288);
  pack_bf16<<<512, 256, 0, stream>>>(Cmat, Cm_bf, 524288);

  // 1) xz = hidden @ W_in^T                    (4096 x 4096, K=2048)
  wmma_gemm_abt_bf<<<dim3(32, 16, 1), 256, 0, stream>>>(hid_bf, Win_bf, xz,
                                                        4096, 4096, 2048, 0, 0, 0);
  // 2) G[b,c] = C_c @ B_c^T (head-shared)      (256 x 256, K=128) x16
  wmma_gemm_abt_bf<<<dim3(2, 1, 16), 256, 0, stream>>>(Cm_bf, Bm_bf, G,
                                                       256, 256, 128,
                                                       256 * 128, 256 * 128, 256 * 256);
  // 3) Y_diag + states per (b,c,h)   (st overwrites Win_bf region: GEMM-1 done)
  intra_chunk<<<1024, 256, 0, stream>>>(xz, G, Bmat, A_log, dt_b, Y, st);
  // 4) chunk recurrence -> prev_states (pv overwrites hid_bf region)
  chunk_scan<<<2048, 256, 0, stream>>>(st, A_log, dt_b, pv);
  // 5) Y_off + D*x + sigmoid gating -> bf16
  inter_chunk<<<1024, 256, 0, stream>>>(xz, Cmat, pv, A_log, dt_b, D_par, Y, yg_bf);
  // 6) out = y @ W_out^T                       (4096 x 2048, K=2048)
  wmma_gemm_abt_bf<<<dim3(32, 8, 1), 256, 0, stream>>>(yg_bf, Wout_bf, out,
                                                       4096, 2048, 2048, 0, 0, 0);
}
